// E1MultiHead_31129922961750
// MI455X (gfx1250) — compile-verified, matrix operand order checked by reading, not run on hardware
//
#include <hip/hip_runtime.h>
#include <math.h>

// Problem constants (match reference)
#define BB     4
#define TT     2048
#define DMODEL 1024
#define NHEAD  16
#define NDIM   64
#define DINNER 1024          // NHEAD * NDIM
#define MTOT   (BB * TT)     // 8192 rows

typedef __attribute__((ext_vector_type(16))) __bf16 v16bf;
typedef __attribute__((ext_vector_type(8)))  __bf16 v8bf;
typedef __attribute__((ext_vector_type(8)))  float  v8f;

// ---------------- helpers ----------------

__device__ __forceinline__ __bf16 f2bf(float f) {
  union { float f; unsigned u; } a; a.f = f;
  unsigned r = a.u + 0x7FFFu + ((a.u >> 16) & 1u);   // round-to-nearest-even
  union { unsigned short s; __bf16 b; } o; o.s = (unsigned short)(r >> 16);
  return o.b;
}

__device__ __forceinline__ float fast_rcp(float x) {
#if __has_builtin(__builtin_amdgcn_rcpf)
  return __builtin_amdgcn_rcpf(x);
#else
  return 1.0f / x;
#endif
}

__device__ __forceinline__ float fast_tanh(float x) {
#if __has_builtin(__builtin_amdgcn_tanhf)
  return __builtin_amdgcn_tanhf(x);      // v_tanh_f32 (gfx1250 trans op)
#else
  return tanhf(x);
#endif
}

__device__ __forceinline__ float fast_silu(float v) {
  return v * fast_rcp(1.0f + __expf(-v));
}

__device__ __forceinline__ v8f v8f_zero() {
  v8f z;
#pragma unroll
  for (int i = 0; i < 8; ++i) z[i] = 0.0f;
  return z;
}

__device__ __forceinline__ v16bf v16bf_zero() {
  v16bf z; __bf16 b0 = f2bf(0.0f);
#pragma unroll
  for (int i = 0; i < 16; ++i) z[i] = b0;
  return z;
}

__device__ __forceinline__ v8f wmma_bf16(v16bf a, v16bf b, v8f c) {
  // D = A(16x32 bf16) * B(32x16 bf16) + C(16x16 f32)
  return __builtin_amdgcn_wmma_f32_16x16x32_bf16(false, a, false, b,
                                                 (short)0, c, false, false);
}

// A fragment: 16x32 bf16 tile from row-major [.., lda] at (m0,k0).
// lanes 0-15: M=lane, K pairs {k0..k0+7, k0+16..k0+23}; lanes 16-31: K+8.
__device__ __forceinline__ v16bf load_a16x32(const __bf16* __restrict__ A,
                                             int lda, int m0, int k0, int lane) {
  int m  = m0 + (lane & 15);
  int kb = k0 + ((lane & 16) ? 8 : 0);
  const __bf16* p = A + (size_t)m * lda + kb;
  v16bf a;
#pragma unroll
  for (int i = 0; i < 8; ++i) { a[i] = p[i]; a[8 + i] = p[16 + i]; }
  return a;
}

// B fragment: 32x16 bf16; source is N-major weights W[n, k] (row-major, ld=ldw),
// i.e. B[k][n] = W[n*ldw + k]. Lane holds one column n; K striped like A.
__device__ __forceinline__ v16bf load_b32x16_nmajor(const __bf16* __restrict__ W,
                                                    int ldw, int n0, int k0, int lane) {
  int n  = n0 + (lane & 15);
  int kb = k0 + ((lane & 16) ? 8 : 0);
  const __bf16* p = W + (size_t)n * ldw + kb;
  v16bf b;
#pragma unroll
  for (int i = 0; i < 8; ++i) { b[i] = p[i]; b[8 + i] = p[16 + i]; }
  return b;
}

// Fragment from an n-major bf16 row pointer (16B-aligned, k0 folded in by caller):
// two contiguous 16B chunks selected by the lane's K-half, concatenated.
__device__ __forceinline__ v16bf frag_from_row16(const __bf16* p, int half /*0|1*/) {
  const v8bf* q = (const v8bf*)p;
  v8bf c0 = q[half];        // k = kb .. kb+7
  v8bf c1 = q[2 + half];    // k = 16+kb .. 23+kb
  return __builtin_shufflevector(c0, c1, 0, 1, 2, 3, 4, 5, 6, 7,
                                 8, 9, 10, 11, 12, 13, 14, 15);
}

// ---------------- kernel 1: fp32 -> bf16 convert ----------------

__global__ void cvt_f32_bf16(const float* __restrict__ src,
                             __bf16* __restrict__ dst, int n) {
  int i = blockIdx.x * blockDim.x + threadIdx.x;
  int stride = gridDim.x * blockDim.x;
  for (; i < n; i += stride) dst[i] = f2bf(src[i]);
}

// ---------------- kernel 2: in_proj GEMM + SiLU ----------------
// xz[m, e] = sum_d x[m,d] * W_in[e,d];  e<1024 -> silu -> bf16 silu_x[m,e]
//                                       e>=1024 -> silu -> f32 gate[t,b,e-1024]
__global__ __launch_bounds__(256) void in_proj_kernel(
    const __bf16* __restrict__ xb,      // [8192, 1024]
    const __bf16* __restrict__ winb,    // [2048, 1024] (n-major)
    __bf16* __restrict__ silu_x,        // [8192, 1024]
    float*  __restrict__ gate)          // [T, B, 1024]
{
  int lane = threadIdx.x & 31;
  int wid  = threadIdx.x >> 5;
  int m0 = blockIdx.x * 128 + (wid & 3) * 32;   // 4 waves along M
  int n0 = blockIdx.y * 128 + (wid >> 2) * 64;  // 2 waves along N

  v8f acc[2][4];
#pragma unroll
  for (int i = 0; i < 2; ++i)
#pragma unroll
    for (int j = 0; j < 4; ++j) acc[i][j] = v8f_zero();

  for (int k0 = 0; k0 < DMODEL; k0 += 32) {
    v16bf a0 = load_a16x32(xb, DMODEL, m0,      k0, lane);
    v16bf a1 = load_a16x32(xb, DMODEL, m0 + 16, k0, lane);
#pragma unroll
    for (int j = 0; j < 4; ++j) {
      v16bf bf = load_b32x16_nmajor(winb, DMODEL, n0 + j * 16, k0, lane);
      acc[0][j] = wmma_bf16(a0, bf, acc[0][j]);
      acc[1][j] = wmma_bf16(a1, bf, acc[1][j]);
    }
  }

  int n15 = lane & 15;
  int mh  = (lane & 16) ? 8 : 0;
  if (n0 < DINNER) {
    // x-branch: silu -> bf16 (uniform: block n-tile never straddles 1024)
#pragma unroll
    for (int i = 0; i < 2; ++i)
#pragma unroll
      for (int j = 0; j < 4; ++j)
#pragma unroll
        for (int r = 0; r < 8; ++r) {
          int mg = m0 + i * 16 + mh + r;
          int ng = n0 + j * 16 + n15;
          silu_x[(size_t)mg * DINNER + ng] = f2bf(fast_silu(acc[i][j][r]));
        }
  } else {
    // gate branch: silu -> f32, time-major layout
#pragma unroll
    for (int i = 0; i < 2; ++i)
#pragma unroll
      for (int j = 0; j < 4; ++j)
#pragma unroll
        for (int r = 0; r < 8; ++r) {
          int mg = m0 + i * 16 + mh + r;
          int ng = n0 + j * 16 + n15 - DINNER;
          int bb = mg >> 11;                    // m = b*T + t
          int tt = mg & (TT - 1);
          gate[(size_t)(tt * BB + bb) * DINNER + ng] = fast_silu(acc[i][j][r]);
        }
  }
}

// ---------------- kernel 3: pre_x = silu_x @ W_x (per head) ----------------
__global__ __launch_bounds__(256) void prex_kernel(
    const __bf16* __restrict__ silu_x,  // [8192, 1024]
    const float*  __restrict__ Wx,      // [H, 64, 64]
    float*        __restrict__ px)      // [T, B, 1024]
{
  // W_x[h] staged in LDS transposed to n-major: wxs_t[n*64 + k] (8 KB, 16B rows)
  __shared__ __attribute__((aligned(16))) __bf16 wxs_t[NDIM * NDIM];
  int h = blockIdx.y;
  int tid = threadIdx.x;
  for (int i = tid; i < NDIM * NDIM; i += 256) {
    int k = i >> 6, n = i & 63;
    wxs_t[n * NDIM + k] = f2bf(Wx[h * NDIM * NDIM + i]);
  }
  __syncthreads();

  int lane = tid & 31, wid = tid >> 5;
  int m0 = blockIdx.x * 128 + wid * 16;       // each wave: 16 rows x full N=64
  int n15 = lane & 15;
  int half = (lane & 16) ? 1 : 0;

  v8f acc[4];
#pragma unroll
  for (int j = 0; j < 4; ++j) acc[j] = v8f_zero();

#pragma unroll
  for (int kk = 0; kk < 2; ++kk) {
    v16bf a = load_a16x32(silu_x, DINNER, m0, h * NDIM + kk * 32, lane);
#pragma unroll
    for (int j = 0; j < 4; ++j) {
      v16bf bf = frag_from_row16(wxs_t + (j * 16 + n15) * NDIM + kk * 32, half);
      acc[j] = wmma_bf16(a, bf, acc[j]);
    }
  }

  int mh = (lane & 16) ? 8 : 0;
#pragma unroll
  for (int j = 0; j < 4; ++j)
#pragma unroll
    for (int r = 0; r < 8; ++r) {
      int mg = m0 + mh + r;
      int bb = mg >> 11, tt = mg & (TT - 1);
      px[(size_t)(tt * BB + bb) * DINNER + h * NDIM + j * 16 + n15] = acc[j][r];
    }
}

// ---------------- kernel 4: sequential scan ----------------
// One wave per head. A = h-state (rows 0-3 = batches, rows 4-15 zero-padded),
// B = W_h[h] resident in 8 VGPR fragments. Per step: 8 independent WMMAs
// (K halves overlap in the pipe), v_tanh, gate, then a bf16 LDS round-trip
// (vector ds_load_b128 + shuffle) to transpose D-layout h_new back to A-layout.
__global__ __launch_bounds__(32) void scan_kernel(
    const float* __restrict__ px,     // [T, B, 1024]
    const float* __restrict__ gate,   // [T, B, 1024]
    const float* __restrict__ h0,     // [B, H, N]
    const float* __restrict__ Wh,     // [H, 64, 64]
    const float* __restrict__ bias,   // [H, 64]
    __bf16*      __restrict__ y,      // [8192, 1024] bf16 (h*gate, row m=b*T+t)
    float*       __restrict__ hfin)   // [B, H, N]
{
  __shared__ __attribute__((aligned(16))) __bf16 lds_hb[BB * NDIM]; // 4x64 bf16
  int h    = blockIdx.x;
  int lane = threadIdx.x;
  int n15  = lane & 15;
  int kb   = (lane & 16) ? 8 : 0;
  int half = kb >> 3;                 // 0 or 1
  int m    = n15;                     // A-layout row owned by this lane

  // Preload W_h[h] as 2(K) x 4(N) B-fragments, fp32 -> bf16 on the fly.
  v16bf Bh[2][4];
  const float* w = Wh + h * NDIM * NDIM;
#pragma unroll
  for (int kk = 0; kk < 2; ++kk)
#pragma unroll
    for (int j = 0; j < 4; ++j) {
#pragma unroll
      for (int i = 0; i < 8; ++i) {
        Bh[kk][j][i]     = f2bf(w[(kk * 32 + kb + i)      * NDIM + j * 16 + n15]);
        Bh[kk][j][8 + i] = f2bf(w[(kk * 32 + kb + 16 + i) * NDIM + j * 16 + n15]);
      }
    }

  float bias_j[4];
#pragma unroll
  for (int j = 0; j < 4; ++j) bias_j[j] = bias[h * NDIM + j * 16 + n15];

  // Initial h-state A-fragments from h0 (rows >= 4 stay zero forever).
  v16bf a0 = v16bf_zero(), a1 = v16bf_zero();
  if (m < BB) {
    const float* hp = h0 + (size_t)(m * NHEAD + h) * NDIM;
#pragma unroll
    for (int i = 0; i < 8; ++i) {
      a0[i]     = f2bf(hp[kb + i]);       a0[8 + i] = f2bf(hp[16 + kb + i]);
      a1[i]     = f2bf(hp[32 + kb + i]);  a1[8 + i] = f2bf(hp[48 + kb + i]);
    }
  }

  const int PF = 8;  // prefetch distance (steps)

  for (int t = 0; t < TT; ++t) {
    // Two independent WMMAs per n-tile (K 0-31 and K 32-63); combine in tanh add.
    v8f z0[4], z1[4];
#pragma unroll
    for (int j = 0; j < 4; ++j) z0[j] = wmma_bf16(a0, Bh[0][j], v8f_zero());
#pragma unroll
    for (int j = 0; j < 4; ++j) z1[j] = wmma_bf16(a1, Bh[1][j], v8f_zero());

    const float* pxt = px   + (size_t)t * BB * DINNER + h * NDIM;
    const float* gt  = gate + (size_t)t * BB * DINNER + h * NDIM;

    // Hide HBM latency of future steps behind the serial chain.
    {
      int tp = t + PF < TT ? t + PF : TT - 1;
      size_t off = (size_t)tp * BB * DINNER + h * NDIM
                 + (size_t)(lane & 3) * DINNER + ((lane >> 2) << 3);
      __builtin_prefetch(px + off, 0, 1);     // global_prefetch_b8
      __builtin_prefetch(gate + off, 0, 1);
    }

#pragma unroll
    for (int j = 0; j < 4; ++j) {
#pragma unroll
      for (int r = 0; r < BB; ++r) {
        float pv = pxt[r * DINNER + j * 16 + n15];
        float gv = gt [r * DINNER + j * 16 + n15];
        float hv = fast_tanh(z0[j][r] + z1[j][r] + pv + bias_j[j]);
        if (lane < 16) {                 // D-layout rows 0-3 live in lanes 0-15
          lds_hb[r * NDIM + j * 16 + n15] = f2bf(hv);
          y[(size_t)(r * TT + t) * DINNER + h * NDIM + j * 16 + n15] = f2bf(hv * gv);
          if (t == TT - 1)
            hfin[(size_t)(r * NHEAD + h) * NDIM + j * 16 + n15] = hv;
        }
      }
    }
    __syncthreads();
    // Transpose h_new (bf16 in LDS) back into A-layout: 4x ds_load_b128 + shuffle.
    if (m < BB) {
      const __bf16* hp = lds_hb + m * NDIM;
      a0 = frag_from_row16(hp, half);        // K 0..31
      a1 = frag_from_row16(hp + 32, half);   // K 32..63
    }
    __syncthreads();
  }
}

// ---------------- kernel 5: out_proj GEMM ----------------
// out[m, d] = sum_e y[m,e] * W_out[d,e]
__global__ __launch_bounds__(256) void out_proj_kernel(
    const __bf16* __restrict__ yb,      // [8192, 1024]
    const __bf16* __restrict__ woutb,   // [1024, 1024] (n-major: d rows, e cols)
    float*        __restrict__ out)     // [8192, 1024]
{
  int lane = threadIdx.x & 31;
  int wid  = threadIdx.x >> 5;
  int m0 = blockIdx.x * 128 + (wid & 3) * 32;
  int n0 = blockIdx.y * 128 + (wid >> 2) * 64;

  v8f acc[2][4];
#pragma unroll
  for (int i = 0; i < 2; ++i)
#pragma unroll
    for (int j = 0; j < 4; ++j) acc[i][j] = v8f_zero();

  for (int k0 = 0; k0 < DINNER; k0 += 32) {
    v16bf a0 = load_a16x32(yb, DINNER, m0,      k0, lane);
    v16bf a1 = load_a16x32(yb, DINNER, m0 + 16, k0, lane);
#pragma unroll
    for (int j = 0; j < 4; ++j) {
      v16bf bf = load_b32x16_nmajor(woutb, DINNER, n0 + j * 16, k0, lane);
      acc[0][j] = wmma_bf16(a0, bf, acc[0][j]);
      acc[1][j] = wmma_bf16(a1, bf, acc[1][j]);
    }
  }

  int n15 = lane & 15;
  int mh  = (lane & 16) ? 8 : 0;
#pragma unroll
  for (int i = 0; i < 2; ++i)
#pragma unroll
    for (int j = 0; j < 4; ++j)
#pragma unroll
      for (int r = 0; r < 8; ++r) {
        int mg = m0 + i * 16 + mh + r;
        int ng = n0 + j * 16 + n15;
        out[(size_t)mg * DMODEL + ng] = acc[i][j][r];
      }
}

// ---------------- host entry ----------------

extern "C" void kernel_launch(void* const* d_in, const int* in_sizes, int n_in,
                              void* d_out, int out_size, void* d_ws, size_t ws_size,
                              hipStream_t stream) {
  (void)in_sizes; (void)n_in; (void)out_size; (void)ws_size;

  const float* x     = (const float*)d_in[0];  // [B,T,DIM]
  const float* h0    = (const float*)d_in[1];  // [B,H,N]
  const float* W_in  = (const float*)d_in[2];  // [2*DINNER, DIM]
  const float* W_x   = (const float*)d_in[3];  // [H,N,N]
  const float* W_h   = (const float*)d_in[4];  // [H,N,N]
  const float* b     = (const float*)d_in[5];  // [H,N]
  const float* W_out = (const float*)d_in[6];  // [DIM, DINNER]

  float* out  = (float*)d_out;                         // [B,T,DIM]
  float* hfin = out + (size_t)MTOT * DMODEL;           // [B,H,N] (tuple tail)

  char* ws = (char*)d_ws;
  __bf16* xb    = (__bf16*)(ws);                        // 16 MB
  __bf16* winb  = (__bf16*)(ws + ((size_t)16  << 20));  //  4 MB
  __bf16* woutb = (__bf16*)(ws + ((size_t)20  << 20));  //  2 MB
  __bf16* sxb   = (__bf16*)(ws + ((size_t)22  << 20));  // 16 MB silu(x_proj)
  float*  gatep = (float*) (ws + ((size_t)38  << 20));  // 32 MB gate [T,B,1024]
  float*  pxp   = (float*) (ws + ((size_t)70  << 20));  // 32 MB pre_x [T,B,1024]
  __bf16* yb    = (__bf16*)(ws + ((size_t)102 << 20));  // 16 MB h*gate bf16

  cvt_f32_bf16<<<1024, 256, 0, stream>>>(x,     xb,    MTOT * DMODEL);
  cvt_f32_bf16<<<512,  256, 0, stream>>>(W_in,  winb,  2 * DINNER * DMODEL);
  cvt_f32_bf16<<<256,  256, 0, stream>>>(W_out, woutb, DMODEL * DINNER);

  in_proj_kernel<<<dim3(MTOT / 128, (2 * DINNER) / 128), 256, 0, stream>>>(
      xb, winb, sxb, gatep);

  prex_kernel<<<dim3(MTOT / 128, NHEAD), 256, 0, stream>>>(sxb, W_x, pxp);

  scan_kernel<<<NHEAD, 32, 0, stream>>>(pxp, gatep, h0, W_h, b, yb, hfin);

  out_proj_kernel<<<dim3(MTOT / 128, DMODEL / 128), 256, 0, stream>>>(
      yb, woutb, out);
}
